// CausalSelfAttention_10591389352413
// MI455X (gfx1250) — matrix operand order repaired and checked by reading, not compiled, not run on hardware
//
#include <hip/hip_runtime.h>

typedef __attribute__((ext_vector_type(16))) __bf16 v16bf;
typedef __attribute__((ext_vector_type(8)))  __bf16 v8bf;
typedef __attribute__((ext_vector_type(8)))  float  v8f;

union BF16x16 { v16bf v; v8bf h[2]; };

__device__ inline unsigned short f2bf(float f) {
  unsigned u = __float_as_uint(f);
  u += 0x7fffu + ((u >> 16) & 1u);           // round-to-nearest-even
  return (unsigned short)(u >> 16);
}

__device__ inline v8f wmma_bf16(v16bf a, v16bf b, v8f c) {
  return __builtin_amdgcn_wmma_f32_16x16x32_bf16(false, a, false, b, (short)0, c,
                                                 false, false);
}

// ---------------- conversion kernels ----------------
__global__ void k_cvt(const float* __restrict__ in, unsigned short* __restrict__ out, int n) {
  int i = blockIdx.x * blockDim.x + threadIdx.x;
  if (i < n) out[i] = f2bf(in[i]);
}

// in [rows][cols] fp32 -> out [cols][rows] bf16 (transpose)
__global__ void k_cvt_t(const float* __restrict__ in, unsigned short* __restrict__ out,
                        int rows, int cols) {
  int i = blockIdx.x * blockDim.x + threadIdx.x;
  if (i < rows * cols) {
    int r = i / cols, c = i % cols;
    out[(size_t)c * rows + r] = f2bf(in[i]);
  }
}

// ---------------- QKV GEMM: [8192,768] x [768,2304] ----------------
// X bf16 row-major [BT][C]; Wt bf16 [N=2304][K=768] (transposed weights).
// One wave computes a 16(M) x 64(N) strip; A fragment reused across 4 B tiles;
// next k-step's fragments prefetched (double buffering). __launch_bounds__(128,1)
// raises the VGPR budget so the scheduler can keep the prefetch in flight.
__global__ void __launch_bounds__(128, 1)
k_qkv(const unsigned short* __restrict__ X,
      const unsigned short* __restrict__ Wt,
      const float* __restrict__ bias,
      unsigned short* __restrict__ Q,
      unsigned short* __restrict__ K,
      unsigned short* __restrict__ Vt) {
  const int lane = threadIdx.x & 31;
  const int wave = threadIdx.x >> 5;
  const int m0 = blockIdx.x << 4;                      // row tile (bt)
  const int n0 = ((blockIdx.y << 2) + wave) << 6;      // 64-wide col strip in [0,2304)
  const int r = lane & 15, h = lane >> 4;

  const unsigned short* arow = X + (size_t)(m0 + r) * 768;
  const unsigned short* b0r = Wt + (size_t)(n0 + r) * 768;
  const unsigned short* b1r = b0r + (size_t)16 * 768;
  const unsigned short* b2r = b0r + (size_t)32 * 768;
  const unsigned short* b3r = b0r + (size_t)48 * 768;

  BF16x16 a, b0, b1, b2, b3;
  a.h[0] = *(const v8bf*)(arow + h * 16);
  a.h[1] = *(const v8bf*)(arow + h * 16 + 8);
  b0.v = *(const v16bf*)(b0r + h * 16);
  b1.v = *(const v16bf*)(b1r + h * 16);
  b2.v = *(const v16bf*)(b2r + h * 16);
  b3.v = *(const v16bf*)(b3r + h * 16);

  v8f c0 = {}, c1 = {}, c2 = {}, c3 = {};
#pragma unroll
  for (int k0 = 0; k0 < 768; k0 += 32) {
    BF16x16 an, b0n, b1n, b2n, b3n;
    const int kn = k0 + 32;
    if (kn < 768) {                       // folds away except at the tail
      an.h[0] = *(const v8bf*)(arow + kn + h * 16);
      an.h[1] = *(const v8bf*)(arow + kn + h * 16 + 8);
      b0n.v = *(const v16bf*)(b0r + kn + h * 16);
      b1n.v = *(const v16bf*)(b1r + kn + h * 16);
      b2n.v = *(const v16bf*)(b2r + kn + h * 16);
      b3n.v = *(const v16bf*)(b3r + kn + h * 16);
    }
    c0 = wmma_bf16(a.v, b0.v, c0);
    c1 = wmma_bf16(a.v, b1.v, c1);
    c2 = wmma_bf16(a.v, b2.v, c2);
    c3 = wmma_bf16(a.v, b3.v, c3);
    a = an; b0 = b0n; b1 = b1n; b2 = b2n; b3 = b3n;
  }

  const int sec = n0 / 768;                            // uniform: 0=q 1=k 2=v
  const int cc = n0 % 768;
  const int head = cc >> 6;                            // uniform per wave
  float bv[4];
#pragma unroll
  for (int j = 0; j < 4; ++j) bv[j] = bias[n0 + j * 16 + r];

#pragma unroll
  for (int i = 0; i < 8; ++i) {
    const int bt = m0 + h * 8 + i;
    const int bb = bt >> 11, t = bt & 2047;
    const size_t bh = (size_t)(bb * 12 + head);
    const float v0 = c0[i] + bv[0], v1 = c1[i] + bv[1];
    const float v2 = c2[i] + bv[2], v3 = c3[i] + bv[3];
    if (sec == 2) {                                    // V^T [bh][hs][t]
      unsigned short* p = Vt + (bh * 64 + r) * 2048 + t;
      p[0 * 16 * 2048] = f2bf(v0);
      p[1 * 16 * 2048] = f2bf(v1);
      p[2 * 16 * 2048] = f2bf(v2);
      p[3 * 16 * 2048] = f2bf(v3);
    } else {                                           // Q/K [bh][t][hs]
      unsigned short* p = (sec == 0 ? Q : K) + (bh * 2048 + t) * 64 + r;
      p[0]  = f2bf(v0);
      p[16] = f2bf(v1);
      p[32] = f2bf(v2);
      p[48] = f2bf(v3);
    }
  }
}

// ---------------- fused causal flash attention ----------------
// Q,K bf16 [BH][T][64]; Vt bf16 [BH][64][T]; Y bf16 [BT][768].
// Block = 4 waves; wave w handles query rows q0..q0+15, q0 = 64*blockIdx.x + 16*w.
__global__ void __launch_bounds__(128, 1)
k_attn(const unsigned short* __restrict__ Qg,
       const unsigned short* __restrict__ Kg,
       const unsigned short* __restrict__ Vg,
       unsigned short* __restrict__ Y) {
  __shared__ __align__(32) unsigned short Pl[4][16 * 32];
  const int lane = threadIdx.x & 31;
  const int wave = threadIdx.x >> 5;
  const int bh = blockIdx.y;
  const int head = bh % 12;
  const int q0 = (blockIdx.x << 6) + (wave << 4);
  const int r = lane & 15, h = lane >> 4;
  const unsigned short* Qb = Qg + (size_t)bh * (2048 * 64);
  const unsigned short* Kb = Kg + (size_t)bh * (2048 * 64);
  const unsigned short* Vb = Vg + (size_t)bh * (64 * 2048);
  unsigned short* P = Pl[wave];

  // preload Q A-fragments (K-chunks 0..31 and 32..63)
  BF16x16 qf0, qf1;
  {
    const unsigned short* qrow = Qb + (size_t)(q0 + r) * 64;
    qf0.h[0] = *(const v8bf*)(qrow + h * 16);
    qf0.h[1] = *(const v8bf*)(qrow + h * 16 + 8);
    qf1.h[0] = *(const v8bf*)(qrow + 32 + h * 16);
    qf1.h[1] = *(const v8bf*)(qrow + 32 + h * 16 + 8);
  }

  v8f o0 = {}, o1 = {}, o2 = {}, o3 = {};      // 16x64 output accumulator
  float m[8], l[8];
#pragma unroll
  for (int i = 0; i < 8; ++i) { m[i] = -3.0e38f; l[i] = 0.f; }
  const float sc = 0.125f * 1.44269504088896f;  // 1/sqrt(64) * log2(e)

  for (int kb = 0; kb < q0 + 16; kb += 32) {    // 32 keys per block (kb <= q0 always)
    // ---- S = Q @ K^T : two 16x16 tiles ----
    v8f s0 = {}, s1 = {};
    {
      BF16x16 b;
      const unsigned short* b0 = Kb + (size_t)(kb + r) * 64;
      b.v = *(const v16bf*)(b0 + h * 16);       s0 = wmma_bf16(qf0.v, b.v, s0);
      b.v = *(const v16bf*)(b0 + 32 + h * 16);  s0 = wmma_bf16(qf1.v, b.v, s0);
      const unsigned short* b1 = Kb + (size_t)(kb + 16 + r) * 64;
      b.v = *(const v16bf*)(b1 + h * 16);       s1 = wmma_bf16(qf0.v, b.v, s1);
      b.v = *(const v16bf*)(b1 + 32 + h * 16);  s1 = wmma_bf16(qf1.v, b.v, s1);
    }
    // ---- V fragments issued early: latency hidden under the softmax VALU work ----
    BF16x16 vb0, vb1, vb2, vb3;
    vb0.v = *(const v16bf*)(Vb + (size_t)(r)      * 2048 + kb + h * 16);
    vb1.v = *(const v16bf*)(Vb + (size_t)(16 + r) * 2048 + kb + h * 16);
    vb2.v = *(const v16bf*)(Vb + (size_t)(32 + r) * 2048 + kb + h * 16);
    vb3.v = *(const v16bf*)(Vb + (size_t)(48 + r) * 2048 + kb + h * 16);
    // ---- online softmax (exp2 domain); lane holds col r, rows h*8+i ----
    const bool needMask = (kb + 31 > q0);
    float p0[8], p1[8];
#pragma unroll
    for (int i = 0; i < 8; ++i) {
      float a = s0[i] * sc, c = s1[i] * sc;
      if (needMask) {
        const int row = q0 + h * 8 + i;
        a = (kb + r      > row) ? -3.0e38f : a;
        c = (kb + 16 + r > row) ? -3.0e38f : c;
      }
      float t = fmaxf(a, c);
      t = fmaxf(t, __shfl_xor(t, 1, 32));
      t = fmaxf(t, __shfl_xor(t, 2, 32));
      t = fmaxf(t, __shfl_xor(t, 4, 32));
      t = fmaxf(t, __shfl_xor(t, 8, 32));       // row-max across the 16-lane half
      const float mn = fmaxf(m[i], t);
      const float cf = exp2f(m[i] - mn);
      m[i] = mn;
      const float e0 = exp2f(a - mn), e1 = exp2f(c - mn);
      p0[i] = e0; p1[i] = e1;
      float rs = e0 + e1;
      rs += __shfl_xor(rs, 1, 32);
      rs += __shfl_xor(rs, 2, 32);
      rs += __shfl_xor(rs, 4, 32);
      rs += __shfl_xor(rs, 8, 32);
      l[i] = l[i] * cf + rs;
      o0[i] *= cf; o1[i] *= cf; o2[i] *= cf; o3[i] *= cf;
    }
    // ---- re-layout P (C-tile layout) -> A-fragment via per-wave LDS bounce ----
#pragma unroll
    for (int i = 0; i < 8; ++i) {
      const int row = h * 8 + i;
      P[row * 32 + r]      = f2bf(p0[i]);
      P[row * 32 + 16 + r] = f2bf(p1[i]);
    }
    asm volatile("s_wait_dscnt 0x0" ::: "memory");  // LDS is in-order per wave
    BF16x16 pa;
    pa.h[0] = *(const v8bf*)(P + r * 32 + h * 16);
    pa.h[1] = *(const v8bf*)(P + r * 32 + h * 16 + 8);
    // ---- O += P @ V ----
    o0 = wmma_bf16(pa.v, vb0.v, o0);
    o1 = wmma_bf16(pa.v, vb1.v, o1);
    o2 = wmma_bf16(pa.v, vb2.v, o2);
    o3 = wmma_bf16(pa.v, vb3.v, o3);
  }
  // ---- normalize and write y[b][t][head*64+hs] as bf16 ----
  const int bb = bh / 12;
  unsigned short* yb = Y + (size_t)bb * 2048 * 768;
#pragma unroll
  for (int i = 0; i < 8; ++i) {
    const float inv = 1.0f / l[i];
    const int t = q0 + h * 8 + i;
    unsigned short* yrow = yb + (size_t)t * 768 + head * 64 + r;
    yrow[0]  = f2bf(o0[i] * inv);
    yrow[16] = f2bf(o1[i] * inv);
    yrow[32] = f2bf(o2[i] * inv);
    yrow[48] = f2bf(o3[i] * inv);
  }
}

// ---------------- output projection: [8192,768] x [768,768] + bias -> fp32 ----------------
// One wave computes a 16(M) x 64(N) strip; double-buffered fragments.
__global__ void __launch_bounds__(128, 1)
k_proj(const unsigned short* __restrict__ Ybf,
       const unsigned short* __restrict__ Wt,
       const float* __restrict__ bias,
       float* __restrict__ out) {
  const int lane = threadIdx.x & 31;
  const int wave = threadIdx.x >> 5;
  const int m0 = blockIdx.x << 4;
  const int n0 = ((blockIdx.y << 2) + wave) << 6;      // 64-wide strip in [0,768)
  const int r = lane & 15, h = lane >> 4;
  const unsigned short* arow = Ybf + (size_t)(m0 + r) * 768;
  const unsigned short* b0r = Wt + (size_t)(n0 + r) * 768;
  const unsigned short* b1r = b0r + (size_t)16 * 768;
  const unsigned short* b2r = b0r + (size_t)32 * 768;
  const unsigned short* b3r = b0r + (size_t)48 * 768;

  BF16x16 a, b0, b1, b2, b3;
  a.h[0] = *(const v8bf*)(arow + h * 16);
  a.h[1] = *(const v8bf*)(arow + h * 16 + 8);
  b0.v = *(const v16bf*)(b0r + h * 16);
  b1.v = *(const v16bf*)(b1r + h * 16);
  b2.v = *(const v16bf*)(b2r + h * 16);
  b3.v = *(const v16bf*)(b3r + h * 16);

  v8f c0 = {}, c1 = {}, c2 = {}, c3 = {};
#pragma unroll
  for (int k0 = 0; k0 < 768; k0 += 32) {
    BF16x16 an, b0n, b1n, b2n, b3n;
    const int kn = k0 + 32;
    if (kn < 768) {
      an.h[0] = *(const v8bf*)(arow + kn + h * 16);
      an.h[1] = *(const v8bf*)(arow + kn + h * 16 + 8);
      b0n.v = *(const v16bf*)(b0r + kn + h * 16);
      b1n.v = *(const v16bf*)(b1r + kn + h * 16);
      b2n.v = *(const v16bf*)(b2r + kn + h * 16);
      b3n.v = *(const v16bf*)(b3r + kn + h * 16);
    }
    c0 = wmma_bf16(a.v, b0.v, c0);
    c1 = wmma_bf16(a.v, b1.v, c1);
    c2 = wmma_bf16(a.v, b2.v, c2);
    c3 = wmma_bf16(a.v, b3.v, c3);
    a = an; b0 = b0n; b1 = b1n; b2 = b2n; b3 = b3n;
  }
  float bv[4];
#pragma unroll
  for (int j = 0; j < 4; ++j) bv[j] = bias[n0 + j * 16 + r];
#pragma unroll
  for (int i = 0; i < 8; ++i) {
    float* orow = out + (size_t)(m0 + h * 8 + i) * 768 + n0 + r;
    orow[0]  = c0[i] + bv[0];
    orow[16] = c1[i] + bv[1];
    orow[32] = c2[i] + bv[2];
    orow[48] = c3[i] + bv[3];
  }
}

// ---------------- host launcher ----------------
extern "C" void kernel_launch(void* const* d_in, const int* in_sizes, int n_in,
                              void* d_out, int out_size, void* d_ws, size_t ws_size,
                              hipStream_t stream) {
  const float* x      = (const float*)d_in[0];
  const float* W_attn = (const float*)d_in[1];
  const float* b_attn = (const float*)d_in[2];
  const float* W_proj = (const float*)d_in[3];
  const float* b_proj = (const float*)d_in[4];
  float* out = (float*)d_out;

  char* ws = (char*)d_ws;
  size_t off = 0;
  auto take = [&](size_t elems) {
    unsigned short* p = (unsigned short*)(ws + off);
    off = (off + elems * 2 + 255) & ~(size_t)255;
    return p;
  };
  unsigned short* xbf = take((size_t)8192 * 768);      // x bf16
  unsigned short* wat = take((size_t)2304 * 768);      // W_attn^T bf16
  unsigned short* wpt = take((size_t)768 * 768);       // W_proj^T bf16
  unsigned short* qb  = take((size_t)48 * 2048 * 64);  // Q [bh][t][hs]
  unsigned short* kb  = take((size_t)48 * 2048 * 64);  // K [bh][t][hs]
  unsigned short* vt  = take((size_t)48 * 2048 * 64);  // V^T [bh][hs][t]
  unsigned short* yb  = take((size_t)8192 * 768);      // attention output bf16

  int n = 8192 * 768;
  k_cvt<<<(n + 255) / 256, 256, 0, stream>>>(x, xbf, n);
  n = 768 * 2304;
  k_cvt_t<<<(n + 255) / 256, 256, 0, stream>>>(W_attn, wat, 768, 2304);
  n = 768 * 768;
  k_cvt_t<<<(n + 255) / 256, 256, 0, stream>>>(W_proj, wpt, 768, 768);

  k_qkv <<<dim3(512, 9), 128, 0, stream>>>(xbf, wat, b_attn, qb, kb, vt);
  k_attn<<<dim3(32, 48), 128, 0, stream>>>(qb, kb, vt, yb);
  k_proj<<<dim3(512, 3), 128, 0, stream>>>(yb, wpt, b_proj, out);
}
